// MambaImgClassifier_9337258902064
// MI455X (gfx1250) — matrix operand order repaired and checked by reading, not compiled, not run on hardware
//
#include <hip/hip_runtime.h>
#include <hip/hip_bf16.h>
#include <math.h>

#define P_     16
#define IMG_   224
#define CH_    3
#define DM_    512
#define NL_    6
#define DI_    1024      // 2*DM
#define NS_    16        // state dim N
#define KC_    4         // conv kernel
#define RR_    32        // dt rank
#define BATCH_ 32
#define GRID_  14        // IMG/P
#define L_     196       // GRID*GRID
#define PD_    768       // CH*P*P
#define BL_    (BATCH_*L_)   // 6272 rows

#define KCH   128        // K-chunk staged per TDM transfer
#define BROW  72         // padded LDS row stride in f16 (128B data + 16B pad)

typedef __attribute__((ext_vector_type(16))) _Float16 v16h;
typedef __attribute__((ext_vector_type(8)))  float    v8f;
typedef __attribute__((ext_vector_type(4)))  unsigned int u32x4;
typedef __attribute__((ext_vector_type(4)))  int i32x4;
typedef __attribute__((ext_vector_type(8)))  int i32x8;

#define HAS_TDM __has_builtin(__builtin_amdgcn_tensor_load_to_lds)

__device__ inline int imin(int a, int b) { return a < b ? a : b; }

#if HAS_TDM
// ---------------------------------------------------------------------------
// Issue one TDM transfer: rows x 64 f16 tile from global (row stride =
// nstride elements) into LDS at byte offset ldsOff, with 16B pad per row
// (LDS row stride = 144B = BROW f16). Tracked by TENSORcnt.
// D# layout per CDNA5 ISA ch.8 (group0: count/lds/global/type,
// group1: data_size, dims, strides, pad config). 2D tensor -> groups 2/3 zero.
// ---------------------------------------------------------------------------
__device__ inline void tdm_load_tile(const _Float16* gsrc, unsigned ldsOff,
                                     int rows, int nstride) {
    unsigned long long ga = (unsigned long long)(size_t)gsrc;
    u32x4 g0;
    g0.x = 1u;                                             // count=1 (valid)
    g0.y = ldsOff;                                         // lds_addr [63:32]
    g0.z = (unsigned)ga;                                   // global_addr[31:0]
    g0.w = (unsigned)((ga >> 32) & 0x01FFFFFFu) | (2u << 30); // [56:32]+type=2
    i32x8 g1;
    unsigned td0 = 64u;                 // tensor_dim0 (elements)
    unsigned td1 = (unsigned)rows;      // tensor_dim1
    // data_size=1 (2B), pad_enable, pad_interval=32DW (code 4), pad_amount=4DW (code 3)
    g1[0] = (int)(0x00010000u | (1u << 20) | (4u << 22) | (3u << 25));
    g1[1] = (int)((td0 & 0xffffu) << 16);                  // tensor_dim0 lo16
    g1[2] = (int)(((td0 >> 16) & 0xffffu) | ((td1 & 0xffffu) << 16));
    g1[3] = (int)(((td1 >> 16) & 0xffffu) | (64u << 16));  // tile_dim0 = 64
    g1[4] = (int)(rows & 0xffff);                          // tile_dim1 = rows
    g1[5] = (int)(unsigned)nstride;                        // dim0_stride lo32
    g1[6] = 0;
    g1[7] = 0;
    i32x4 gz = {0, 0, 0, 0};
#if defined(__clang_major__) && (__clang_major__ >= 23)
    i32x8 gz8 = {0, 0, 0, 0, 0, 0, 0, 0};
    __builtin_amdgcn_tensor_load_to_lds(g0, g1, gz, gz, gz8, 0);
#else
    __builtin_amdgcn_tensor_load_to_lds(g0, g1, gz, gz, 0);
#endif
}
#endif

// ---------------------------------------------------------------------------
// WMMA GEMM: C[M,N] = A[M,K] * B[K,N] (+bias), f16 in, f32 out.
// B is pre-transposed (Wt = W^T, K x N). Block = 8 waves = 128x64 output.
// Wt K-chunks (128x64) are staged into LDS once per block (TDM async,
// double-buffered) and shared by all 8 waves; each wave computes a 16x64
// strip (4 WMMA accumulators), A fragments streamed via global_load_b128.
// Fragment layouts per ISA 7.12.2 (see round-1 comments).
// ---------------------------------------------------------------------------
__global__ void __launch_bounds__(256)
k_wmma_gemm_tdm(const _Float16* __restrict__ A, int lda,   // M x K
                const _Float16* __restrict__ Wt,           // K x N
                const float* __restrict__ bias,
                float* __restrict__ C, int ldc,
                int M, int N, int Kd) {
    __shared__ _Float16 sB[2][KCH * BROW];   // 2 x 18KB, padded rows
    int tid  = threadIdx.x;
    int lane = tid & 31;
    int wv   = tid >> 5;
    int ngn  = N >> 6;
    int mblk = blockIdx.x / ngn;
    int ng   = blockIdx.x - mblk * ngn;
    int n0   = ng << 6;
    int mt   = mblk * 8 + wv;

    int arow  = lane & 15;
    int kbase = (lane < 16) ? 0 : 8;
    const _Float16* arowp = A + (size_t)(mt * 16 + arow) * lda;

    int nch = (Kd + KCH - 1) / KCH;
    unsigned ldsBase  = (unsigned)(size_t)(&sB[0][0]);
    unsigned bufBytes = KCH * BROW * 2;

#if HAS_TDM
    if (wv == 0)
        tdm_load_tile(Wt + n0, ldsBase, imin(Kd, KCH), N);
#endif

    v8f acc0 = {}, acc1 = {}, acc2 = {}, acc3 = {};
    for (int j = 0; j < nch; ++j) {
        int k0 = j * KCH;
        int kc = imin(KCH, Kd - k0);
#if HAS_TDM
        if (wv == 0) {
            if (j + 1 < nch) {
                tdm_load_tile(Wt + (size_t)(k0 + KCH) * N + n0,
                              ldsBase + (unsigned)((j + 1) & 1) * bufBytes,
                              imin(KCH, Kd - k0 - KCH), N);
                __builtin_amdgcn_s_wait_tensorcnt((short)1);  // chunk j done
            } else {
                __builtin_amdgcn_s_wait_tensorcnt((short)0);
            }
        }
        __syncthreads();                 // buffer j visible to all waves
#else
        __syncthreads();                 // previous compute done
        for (int idx = tid; idx < kc * 32; idx += 256) {
            int r = idx >> 5, w = idx & 31;
            ((unsigned*)&sB[j & 1][r * BROW])[w] =
                ((const unsigned*)(Wt + (size_t)(k0 + r) * N + n0))[w];
        }
        __syncthreads();
#endif
        const _Float16* sb = &sB[j & 1][0];
        for (int kk = 0; kk < kc; kk += 32) {
            const _Float16* ap = arowp + k0 + kk + kbase;
            v16h a;
#pragma unroll
            for (int i = 0; i < 8; ++i) {
                a[i]     = ap[i];        // K = kbase + i
                a[8 + i] = ap[16 + i];   // K = kbase + 16 + i
            }
            __builtin_prefetch(ap + 32, 0, 1);   // next k-step A line
            const _Float16* wp = sb + kk * BROW + (lane & 15);  // column n = lane%16
            v16h b0, b1, b2, b3;
#pragma unroll
            for (int i = 0; i < 16; ++i) {
                int kr = (i < 8 ? kbase + i : kbase + 8 + i) * BROW;
                b0[i] = wp[kr];
                b1[i] = wp[kr + 16];
                b2[i] = wp[kr + 32];
                b3[i] = wp[kr + 48];
            }
            acc0 = __builtin_amdgcn_wmma_f32_16x16x32_f16(false, a, false, b0,
                                                          (short)0, acc0, false, false);
            acc1 = __builtin_amdgcn_wmma_f32_16x16x32_f16(false, a, false, b1,
                                                          (short)0, acc1, false, false);
            acc2 = __builtin_amdgcn_wmma_f32_16x16x32_f16(false, a, false, b2,
                                                          (short)0, acc2, false, false);
            acc3 = __builtin_amdgcn_wmma_f32_16x16x32_f16(false, a, false, b3,
                                                          (short)0, acc3, false, false);
        }
        __syncthreads();                 // all waves done with buffer j
    }

    int crow = mt * 16 + ((lane < 16) ? 0 : 8);
    int ccol = n0 + (lane & 15);
    float* crp = C + (size_t)crow * ldc + ccol;
    float bv0 = bias ? bias[ccol]      : 0.0f;
    float bv1 = bias ? bias[ccol + 16] : 0.0f;
    float bv2 = bias ? bias[ccol + 32] : 0.0f;
    float bv3 = bias ? bias[ccol + 48] : 0.0f;
#pragma unroll
    for (int r = 0; r < 8; ++r) {
        float* cp = crp + (size_t)r * ldc;
        cp[0]  = acc0[r] + bv0;
        cp[16] = acc1[r] + bv1;
        cp[32] = acc2[r] + bv2;
        cp[48] = acc3[r] + bv3;
    }
}

// ---------------------------------------------------------------------------
// Weight convert f32 -> f16 with transpose: W (N x K) -> Wt (K x N).
// ---------------------------------------------------------------------------
__global__ void k_cvt_transpose(const float* __restrict__ W,
                                _Float16* __restrict__ Wt, int N, int K) {
    int idx = blockIdx.x * blockDim.x + threadIdx.x;
    if (idx >= N * K) return;
    int n = idx / K, k = idx - n * K;
    Wt[(size_t)k * N + n] = (_Float16)W[idx];
}

// plain f32 -> f16 (activations)
__global__ void k_f32_to_f16(const float* __restrict__ in,
                             _Float16* __restrict__ out, int n) {
    int i = blockIdx.x * blockDim.x + threadIdx.x;
    if (i < n) out[i] = (_Float16)in[i];
}

// ---------------------------------------------------------------------------
// Patchify + LayerNorm(PD=768); one block (256 thr) per (b,l) row; f16 out.
// ---------------------------------------------------------------------------
__global__ void k_patchify_ln(const float* __restrict__ x,
                              const float* __restrict__ g,
                              const float* __restrict__ beta,
                              _Float16* __restrict__ out_h) {
    int row = blockIdx.x;               // b*L + l
    int b = row / L_, l = row - b * L_;
    int gh = l / GRID_, gw = l - gh * GRID_;
    int tid = threadIdx.x;

    float vals[3];
    float s = 0.f, sq = 0.f;
#pragma unroll
    for (int j = 0; j < 3; ++j) {
        int pd = tid + j * 256;
        int c  = pd / (P_ * P_);
        int r  = pd - c * (P_ * P_);
        int p1 = r / P_, p2 = r - p1 * P_;
        float v = x[((size_t)(b * CH_ + c) * IMG_ + gh * P_ + p1) * IMG_ +
                    gw * P_ + p2];
        vals[j] = v; s += v; sq += v * v;
    }
    __shared__ float ssum[256], ssq[256];
    ssum[tid] = s; ssq[tid] = sq;
    __syncthreads();
    for (int st = 128; st > 0; st >>= 1) {
        if (tid < st) { ssum[tid] += ssum[tid + st]; ssq[tid] += ssq[tid + st]; }
        __syncthreads();
    }
    float mean = ssum[0] * (1.0f / PD_);
    float var  = ssq[0] * (1.0f / PD_) - mean * mean;
    float rstd = rsqrtf(var + 1e-5f);
#pragma unroll
    for (int j = 0; j < 3; ++j) {
        int pd = tid + j * 256;
        float v = (vals[j] - mean) * rstd * g[pd] + beta[pd];
        out_h[(size_t)row * PD_ + pd] = (_Float16)v;
    }
}

// ---------------------------------------------------------------------------
// out = LayerNorm(xin (+ res)) over D=512; writes f32 + f16 shadow.
// One block (256 thr) per row; safe for in-place res==out.
// ---------------------------------------------------------------------------
__global__ void k_add_ln512(const float* __restrict__ xin,
                            const float* __restrict__ res,
                            const float* __restrict__ g,
                            const float* __restrict__ beta,
                            float* __restrict__ out,
                            _Float16* __restrict__ out_h) {
    int row = blockIdx.x;
    int tid = threadIdx.x;
    float vals[2];
    float s = 0.f, sq = 0.f;
#pragma unroll
    for (int j = 0; j < 2; ++j) {
        int d = tid + j * 256;
        float v = xin[(size_t)row * DM_ + d];
        if (res) v += res[(size_t)row * DM_ + d];
        vals[j] = v; s += v; sq += v * v;
    }
    __shared__ float ssum[256], ssq[256];
    ssum[tid] = s; ssq[tid] = sq;
    __syncthreads();
    for (int st = 128; st > 0; st >>= 1) {
        if (tid < st) { ssum[tid] += ssum[tid + st]; ssq[tid] += ssq[tid + st]; }
        __syncthreads();
    }
    float mean = ssum[0] * (1.0f / DM_);
    float var  = ssq[0] * (1.0f / DM_) - mean * mean;
    float rstd = rsqrtf(var + 1e-5f);
#pragma unroll
    for (int j = 0; j < 2; ++j) {
        int d = tid + j * 256;
        float v = (vals[j] - mean) * rstd * g[d] + beta[d];
        out[(size_t)row * DM_ + d]   = v;
        out_h[(size_t)row * DM_ + d] = (_Float16)v;
    }
}

// ---------------------------------------------------------------------------
// Causal depthwise conv (K=4) + bias + SiLU. xs = xz[:, 0:DI].
// ---------------------------------------------------------------------------
__global__ void k_conv_silu(const float* __restrict__ xz,     // BL x 2DI
                            const float* __restrict__ convW,  // DI x 4
                            const float* __restrict__ convB,  // DI
                            float* __restrict__ xc,
                            _Float16* __restrict__ xc_h) {
    int idx = blockIdx.x * blockDim.x + threadIdx.x;   // over BL*DI
    if (idx >= BL_ * DI_) return;
    int d   = idx & (DI_ - 1);
    int row = idx >> 10;                // b*L + l
    int l   = row % L_;
    float acc = convB[d];
#pragma unroll
    for (int k = 0; k < KC_; ++k) {
        int li = l + k - (KC_ - 1);
        if (li >= 0)
            acc += convW[d * KC_ + k] *
                   xz[(size_t)(row + li - l) * (2 * DI_) + d];
    }
    float sv = acc / (1.0f + expf(-acc));   // silu
    xc[idx]   = sv;
    xc_h[idx] = (_Float16)sv;
}

// ---------------------------------------------------------------------------
// dt = softplus(dt_pre + dt_bias), in place.
// ---------------------------------------------------------------------------
__global__ void k_bias_softplus(float* __restrict__ dt,
                                const float* __restrict__ dtB) {
    int idx = blockIdx.x * blockDim.x + threadIdx.x;
    if (idx >= BL_ * DI_) return;
    int d = idx & (DI_ - 1);
    float v = dt[idx] + dtB[d];
    dt[idx] = (v > 20.0f) ? v : log1pf(expf(v));
}

// ---------------------------------------------------------------------------
// Selective scan: one thread per (b,d); h[16] in registers, L=196 steps.
// Fuses D*x skip and SiLU(z) gate; emits f16 for out_proj GEMM.
// ---------------------------------------------------------------------------
__global__ void k_scan(const float* __restrict__ xc,     // BL x DI
                       const float* __restrict__ dt,     // BL x DI
                       const float* __restrict__ xdbl,   // BL x 64 (B:32..47, C:48..63)
                       const float* __restrict__ A_log,  // DI x 16
                       const float* __restrict__ Dp,     // DI
                       const float* __restrict__ xz,     // BL x 2DI (z at +DI)
                       _Float16* __restrict__ y_h) {
    int d = blockIdx.x * blockDim.x + threadIdx.x;
    int b = blockIdx.y;
    if (d >= DI_) return;
    float A[NS_], h[NS_];
#pragma unroll
    for (int n = 0; n < NS_; ++n) {
        A[n] = -expf(A_log[(size_t)d * NS_ + n]);
        h[n] = 0.0f;
    }
    float Dv = Dp[d];
    for (int l = 0; l < L_; ++l) {
        size_t row = (size_t)b * L_ + l;
        float dtv = dt[row * DI_ + d];
        float xv  = xc[row * DI_ + d];
        const float* bc = xdbl + row * 64;
        float y = 0.0f;
#pragma unroll
        for (int n = 0; n < NS_; ++n) {
            float dA = expf(dtv * A[n]);
            h[n] = dA * h[n] + dtv * bc[32 + n] * xv;
            y += h[n] * bc[48 + n];
        }
        y += xv * Dv;
        float zv = xz[row * (2 * DI_) + DI_ + d];
        y *= zv / (1.0f + expf(-zv));       // * silu(z)
        y_h[row * DI_ + d] = (_Float16)y;
    }
}

// ---------------------------------------------------------------------------
// Mean-pool over L, then small head GEMV.
// ---------------------------------------------------------------------------
__global__ void k_pool(const float* __restrict__ t, float* __restrict__ pooled) {
    int idx = blockIdx.x * blockDim.x + threadIdx.x;   // over BATCH*DM
    if (idx >= BATCH_ * DM_) return;
    int b = idx / DM_, dmi = idx - b * DM_;
    float s = 0.0f;
    for (int l = 0; l < L_; ++l)
        s += t[((size_t)b * L_ + l) * DM_ + dmi];
    pooled[idx] = s * (1.0f / L_);
}

__global__ void k_head(const float* __restrict__ pooled,
                       const float* __restrict__ Wh,
                       const float* __restrict__ bh,
                       float* __restrict__ out) {
    int idx = blockIdx.x * blockDim.x + threadIdx.x;   // over BATCH*10
    if (idx >= BATCH_ * 10) return;
    int b = idx / 10, j = idx - b * 10;
    float s = bh[j];
    for (int k = 0; k < DM_; ++k)
        s += pooled[(size_t)b * DM_ + k] * Wh[(size_t)j * DM_ + k];
    out[idx] = s;
}

// ---------------------------------------------------------------------------
extern "C" void kernel_launch(void* const* d_in, const int* in_sizes, int n_in,
                              void* d_out, int out_size, void* d_ws, size_t ws_size,
                              hipStream_t stream) {
    const float* x         = (const float*)d_in[0];
    const float* ln0_g     = (const float*)d_in[1];
    const float* ln0_b     = (const float*)d_in[2];
    const float* W_emb     = (const float*)d_in[3];
    const float* b_emb     = (const float*)d_in[4];
    const float* ln1_g     = (const float*)d_in[5];
    const float* ln1_b     = (const float*)d_in[6];
    const float* in_proj_w = (const float*)d_in[7];    // NL x 2DI x DM
    const float* conv_w    = (const float*)d_in[8];    // NL x DI x 4
    const float* conv_b    = (const float*)d_in[9];    // NL x DI
    const float* x_proj_w  = (const float*)d_in[10];   // NL x 64 x DI
    const float* dt_proj_w = (const float*)d_in[11];   // NL x DI x 32
    const float* dt_proj_b = (const float*)d_in[12];   // NL x DI
    const float* A_log     = (const float*)d_in[13];   // NL x DI x 16
    const float* D_p       = (const float*)d_in[14];   // NL x DI
    const float* out_proj_w= (const float*)d_in[15];   // NL x DM x DI
    const float* blk_ln_g  = (const float*)d_in[16];   // NL x DM
    const float* blk_ln_b  = (const float*)d_in[17];   // NL x DM
    const float* W_head    = (const float*)d_in[18];   // 10 x DM
    const float* b_head    = (const float*)d_in[19];   // 10
    float* out = (float*)d_out;

    // workspace carve-out (single stream -> ordered reuse is safe)
    char* base = (char*)d_ws;
    size_t off = 0;
    auto alloc = [&](size_t bytes) -> void* {
        void* p = base + off;
        off += (bytes + 255) & ~(size_t)255;
        return p;
    };
    _Float16* Xp_h   = (_Float16*)alloc((size_t)BL_ * PD_ * 2);
    float*    t      = (float*)   alloc((size_t)BL_ * DM_ * 4);
    _Float16* t_h    = (_Float16*)alloc((size_t)BL_ * DM_ * 2);
    float*    xz     = (float*)   alloc((size_t)BL_ * 2 * DI_ * 4);
    float*    xc     = (float*)   alloc((size_t)BL_ * DI_ * 4);
    _Float16* xc_h   = (_Float16*)alloc((size_t)BL_ * DI_ * 2);
    float*    xdbl   = (float*)   alloc((size_t)BL_ * 64 * 4);
    _Float16* xdbl_h = (_Float16*)alloc((size_t)BL_ * 64 * 2);
    float*    dtb    = (float*)   alloc((size_t)BL_ * DI_ * 4);
    _Float16* y_h    = (_Float16*)alloc((size_t)BL_ * DI_ * 2);
    float*    attn   = (float*)   alloc((size_t)BL_ * DM_ * 4);
    float*    pooled = (float*)   alloc((size_t)BATCH_ * DM_ * 4);
    _Float16* wbuf   = (_Float16*)alloc((size_t)2 * DI_ * DM_ * 2); // largest Wt

    // W (NxK f32, row-major) -> Wt (KxN f16) in wbuf
    auto cvtT = [&](const float* src, int N, int K) {
        int n = N * K;
        k_cvt_transpose<<<(n + 255) / 256, 256, 0, stream>>>(src, wbuf, N, K);
    };
    auto gemm = [&](const _Float16* A, int lda, const float* bias,
                    float* Cm, int ldc, int M, int N, int Kd) {
        int blocks = (M / 128) * (N / 64);       // 8 waves/block, 128x64 tile
        k_wmma_gemm_tdm<<<blocks, 256, 0, stream>>>(A, lda, wbuf, bias, Cm,
                                                    ldc, M, N, Kd);
    };

    // ---- patchify + LN0 -> f16 ----
    k_patchify_ln<<<BL_, 256, 0, stream>>>(x, ln0_g, ln0_b, Xp_h);

    // ---- embedding GEMM (+bias), then LN1 -> t/t_h ----
    cvtT(W_emb, DM_, PD_);
    gemm(Xp_h, PD_, b_emb, attn, DM_, BL_, DM_, PD_);
    k_add_ln512<<<BL_, 256, 0, stream>>>(attn, nullptr, ln1_g, ln1_b, t, t_h);

    const int nEl = BL_ * DI_;
    for (int i = 0; i < NL_; ++i) {
        const float* inW  = in_proj_w  + (size_t)i * 2 * DI_ * DM_;
        const float* cW   = conv_w     + (size_t)i * DI_ * KC_;
        const float* cB   = conv_b     + (size_t)i * DI_;
        const float* xpW  = x_proj_w   + (size_t)i * 64 * DI_;
        const float* dtW  = dt_proj_w  + (size_t)i * DI_ * RR_;
        const float* dtB  = dt_proj_b  + (size_t)i * DI_;
        const float* Alg  = A_log      + (size_t)i * DI_ * NS_;
        const float* Dv   = D_p        + (size_t)i * DI_;
        const float* oW   = out_proj_w + (size_t)i * DM_ * DI_;
        const float* bg   = blk_ln_g   + (size_t)i * DM_;
        const float* bb   = blk_ln_b   + (size_t)i * DM_;

        // xz = t @ inW^T : (6272,512)x(512,2048)
        cvtT(inW, 2 * DI_, DM_);
        gemm(t_h, DM_, nullptr, xz, 2 * DI_, BL_, 2 * DI_, DM_);

        // causal depthwise conv + silu
        k_conv_silu<<<(nEl + 255) / 256, 256, 0, stream>>>(xz, cW, cB, xc, xc_h);

        // x_dbl = xc @ xpW^T : (6272,1024)x(1024,64)
        cvtT(xpW, 64, DI_);
        gemm(xc_h, DI_, nullptr, xdbl, 64, BL_, 64, DI_);
        k_f32_to_f16<<<(BL_ * 64 + 255) / 256, 256, 0, stream>>>(xdbl, xdbl_h,
                                                                 BL_ * 64);

        // dt = softplus(x_dbl[:, :32] @ dtW^T + dtB) : (6272,32)x(32,1024)
        cvtT(dtW, DI_, RR_);
        gemm(xdbl_h, 64, nullptr, dtb, DI_, BL_, DI_, RR_);
        k_bias_softplus<<<(nEl + 255) / 256, 256, 0, stream>>>(dtb, dtB);

        // selective scan (+ D skip + silu(z) gate) -> y_h f16
        k_scan<<<dim3(DI_ / 256, BATCH_), 256, 0, stream>>>(
            xc, dtb, xdbl, Alg, Dv, xz, y_h);

        // out = y @ oW^T : (6272,1024)x(1024,512)
        cvtT(oW, DM_, DI_);
        gemm(y_h, DI_, nullptr, attn, DM_, BL_, DM_, DI_);

        // t = LN(out + t)
        k_add_ln512<<<BL_, 256, 0, stream>>>(attn, t, bg, bb, t, t_h);
    }

    // ---- mean pool + head ----
    k_pool<<<(BATCH_ * DM_ + 255) / 256, 256, 0, stream>>>(t, pooled);
    k_head<<<(BATCH_ * 10 + 255) / 256, 256, 0, stream>>>(pooled, W_head,
                                                          b_head, out);
}
